// StageMLPBlock_1382979469441
// MI455X (gfx1250) — matrix operand order repaired and checked by reading, not compiled, not run on hardware
//
#include <hip/hip_runtime.h>
#include <math.h>

#define BATCH 32
#define CH    128
#define HH    56
#define WW    56
#define DD    10
#define HW_   (HH*WW)            // 3136
#define TOK   (BATCH*HH*WW)      // 100352
#define NN    HW_                // 3136 real wnn output cols
#define NP    3200               // padded N (25*128)
#define KP1   (WW*DD)            // 560 real K for wnn gemm
#define KP1P  576                // padded K (18*32)
#define ROWS  (BATCH*HH)         // 1792
#define EC    (4*CH)             // 512
#define ABUFB (128*40*2)         // bytes per LDS A buffer

typedef __bf16 v16bf __attribute__((ext_vector_type(16)));
typedef float  v8f   __attribute__((ext_vector_type(8)));
typedef float  f32x4 __attribute__((ext_vector_type(4)));
typedef unsigned short ushort8 __attribute__((ext_vector_type(8)));
typedef unsigned int   uint32x4 __attribute__((ext_vector_type(4)));
typedef int            int32x4  __attribute__((ext_vector_type(4)));
typedef int            int32x8  __attribute__((ext_vector_type(8)));

union Frag {
    v16bf v;
    ushort8 h[2];
    unsigned short u[16];
};

__device__ __forceinline__ unsigned short f2bf(float x) {
    unsigned int u = __float_as_uint(x);
    unsigned int r = u + 0x7FFFu + ((u >> 16) & 1u);   // round-to-nearest-even
    return (unsigned short)(r >> 16);
}
__device__ __forceinline__ float bf2f(unsigned short s) {
    return __uint_as_float(((unsigned int)s) << 16);
}
__device__ __forceinline__ float gelu_exact(float x) {
    return 0.5f * x * (1.0f + erff(x * 0.70710678118654752f));
}
__device__ __forceinline__ ushort8 cvt8(f32x4 a, f32x4 b) {
    ushort8 r;
#pragma unroll
    for (int q = 0; q < 4; ++q) { r[q] = f2bf(a[q]); r[q + 4] = f2bf(b[q]); }
    return r;
}

// ---------------------------------------------------------------------------
// Tensor Data Mover: load a 128-row x 32-col bf16 tile (row stride = lda
// elements) from global into LDS with 8-element padding per row (row stride
// 40 elements in LDS, matching the sA layout). 2D descriptor, groups 2/3 = 0.
// pad_interval=3 -> pad after every 16 DWORDs (one 32-elem bf16 row);
// pad_amount=3 -> 4 DWORDs (8 bf16) padding.
// ---------------------------------------------------------------------------
__device__ __forceinline__ void tdm_load_tile(const unsigned short* gptr,
                                              int lda, unsigned int lds_off) {
    unsigned long long ga = (unsigned long long)gptr;
    uint32x4 g0;
    g0.x = 1u;                                   // count=1 (user mode)
    g0.y = lds_off;                              // lds_addr (bytes)
    g0.z = (unsigned int)(ga & 0xFFFFFFFFu);     // global_addr[31:0]
    g0.w = (unsigned int)((ga >> 32) & 0x01FFFFFFu) | 0x80000000u; // [56:32]|type=2
    unsigned int dim0 = (unsigned int)lda;       // tensor_dim0 (elements)
    unsigned int dim1 = 1u << 20;                // plenty of rows (no clipping)
    int32x8 g1;
    g1[0] = (int)((1u << 16)        // data_size = 2 bytes
                | (1u << 20)        // pad_enable
                | (3u << 22)        // pad_interval = 16 DWORDs
                | (3u << 25));      // pad_amount  = 4 DWORDs
    g1[1] = (int)((dim0 & 0xFFFFu) << 16);                 // tensor_dim0 lo16
    g1[2] = (int)((dim0 >> 16) | ((dim1 & 0xFFFFu) << 16));// dim0 hi16 | dim1 lo16
    g1[3] = (int)((dim1 >> 16) | (32u << 16));             // dim1 hi16 | tile_dim0=32
    g1[4] = (int)128u;                                     // tile_dim1=128, tile_dim2=0
    g1[5] = (int)dim0;                                     // tensor_dim0_stride lo32
    g1[6] = 0;                                             // stride0 hi16 | stride1 lo16
    g1[7] = 0;
    int32x4 gz = {0, 0, 0, 0};
#if defined(__clang_major__) && (__clang_major__ >= 23)
    int32x8 gz8 = {0, 0, 0, 0, 0, 0, 0, 0};
    __builtin_amdgcn_tensor_load_to_lds(g0, g1, gz, gz, gz8, 0);
#else
    __builtin_amdgcn_tensor_load_to_lds(g0, g1, gz, gz, 0);
#endif
}

// ---------------------------------------------------------------------------
__global__ void k_fill_u16(unsigned short* __restrict__ dst, int n) {
    int i = blockIdx.x * blockDim.x + threadIdx.x;
    if (i < n) dst[i] = 0;
}

__global__ void k_cast_bf16(const float* __restrict__ src,
                            unsigned short* __restrict__ dst, int n) {
    int i = blockIdx.x * blockDim.x + threadIdx.x;
    if (i < n) dst[i] = f2bf(src[i]);
}

// pad-cast wnn: [560][3136] f32 -> [576][3200] bf16 (zeros in pad region)
__global__ void k_cast_pad_wnn(const float* __restrict__ src,
                               unsigned short* __restrict__ dst) {
    int i = blockIdx.x * blockDim.x + threadIdx.x;
    if (i >= KP1P * NP) return;
    int r = i / NP, c = i % NP;
    dst[i] = (r < KP1 && c < NN) ? f2bf(src[(long)r * NN + c]) : (unsigned short)0;
}

// ---------------------------------------------------------------------------
// LayerNorm over C=128; one token per 128-thread block.
// ---------------------------------------------------------------------------
__global__ __launch_bounds__(128) void k_layernorm(
    const float* __restrict__ src, const float* __restrict__ g,
    const float* __restrict__ b, unsigned short* __restrict__ dst,
    int transposed) {
    int t = blockIdx.x;
    int c = threadIdx.x;
    __shared__ float s1[128], s2[128];
    float x;
    if (transposed) {
        int bb = t / HW_, rem = t % HW_;
        x = src[(long)bb * CH * HW_ + (long)c * HW_ + rem];
    } else {
        x = src[(long)t * CH + c];
    }
    s1[c] = x; s2[c] = x * x;
    __syncthreads();
    for (int off = 64; off > 0; off >>= 1) {
        if (c < off) { s1[c] += s1[c + off]; s2[c] += s2[c + off]; }
        __syncthreads();
    }
    float mu  = s1[0] * (1.0f / CH);
    float var = s2[0] * (1.0f / CH) - mu * mu;
    float inv = rsqrtf(var + 1e-5f);
    dst[(long)t * CH + c] = f2bf((x - mu) * inv * g[c] + b[c]);
}

// ---------------------------------------------------------------------------
// p = u @ wnd + bnd for both mixers; writes both reshape orders into padded
// [ROWS][576] buffers (tail cols pre-zeroed by k_fill_u16).
// ---------------------------------------------------------------------------
__global__ void k_p1(const unsigned short* __restrict__ u,
                     const float* __restrict__ hwnd, const float* __restrict__ hbnd,
                     const float* __restrict__ wwnd, const float* __restrict__ wbnd,
                     unsigned short* __restrict__ p1h,
                     unsigned short* __restrict__ p1w) {
    int idx = blockIdx.x * blockDim.x + threadIdx.x;
    if (idx >= TOK * DD) return;
    int t = idx / DD, j = idx % DD;
    const unsigned short* ur = u + (long)t * CH;
    float ah = 0.f, aw = 0.f;
    for (int c = 0; c < CH; ++c) {
        float uv = bf2f(ur[c]);
        ah += uv * hwnd[c * DD + j];
        aw += uv * wwnd[c * DD + j];
    }
    int bb = t / HW_, rem = t % HW_, h = rem / WW, w = rem % WW;
    p1h[((long)(bb * HH + h)) * KP1P + w * DD + j] = f2bf(ah + hbnd[j]);
    p1w[((long)(bb * WW + w)) * KP1P + h * DD + j] = f2bf(aw + wbnd[j]);
}

// ---------------------------------------------------------------------------
// Row softmax over 56 values with padded output [1792][64][64] bf16.
// grid = 1792*64 (pad rows written as zeros). src f32 [1792][3200].
// ---------------------------------------------------------------------------
__global__ __launch_bounds__(64) void k_softmax56(const float* __restrict__ src,
                                                  unsigned short* __restrict__ dst) {
    int bx = blockIdx.x;
    int group = bx >> 6, inner = bx & 63;
    int i = threadIdx.x;
    unsigned short* drow = dst + ((long)group * 64 + inner) * 64;
    if (inner >= 56) { drow[i] = 0; return; }
    __shared__ float sh[64];
    long so = (long)group * NP + (long)inner * 56;
    float v = (i < 56) ? src[so + i] : -1e30f;
    sh[i] = v; __syncthreads();
    for (int off = 32; off > 0; off >>= 1) {
        if (i < off) sh[i] = fmaxf(sh[i], sh[i + off]);
        __syncthreads();
    }
    float mx = sh[0];
    __syncthreads();
    float e = (i < 56) ? expf(v - mx) : 0.f;
    sh[i] = e; __syncthreads();
    for (int off = 32; off > 0; off >>= 1) {
        if (i < off) sh[i] += sh[i + off];
        __syncthreads();
    }
    float inv = 1.0f / sh[0];
    drow[i] = (i < 56) ? f2bf(e * inv) : (unsigned short)0;
}

// ---------------------------------------------------------------------------
// Fast tiled bf16 WMMA GEMM, all dims tile-aligned (M%128,N%128,K%32==0).
// AMODE 0: A is bf16; A tiles come in via the Tensor Data Mover into double-
//          buffered LDS (next tile's TDM overlaps current tile's WMMAs).
// AMODE 1: A is f32; register-staged and converted to bf16 on the LDS write
//          (fuses the f32->bf16 cast pass into the GEMM).
// B is always register double-buffered and stored transposed in LDS.
// RES: 0 none; 1 f32 row-major residual; 2 f32 [B,C,H,W] transposed residual
// OUT: 0 f32 row-major; 1 bf16 row-major; 2 f32 [B,C,H,W] transposed
// ACT: 1 = exact GELU.  NGUARD: bias columns limited to Nbias (padded N).
// ---------------------------------------------------------------------------
template <int RES, int OUT, int ACT, bool NGUARD, int AMODE>
__global__ __launch_bounds__(256) void k_gemm(
    const void* __restrict__ Aptr, int lda,
    const unsigned short* __restrict__ B, int ldb,
    int K, int Nbias,
    const float* __restrict__ bias,
    const float* __restrict__ Res, int ldres,
    float* __restrict__ Cf, unsigned short* __restrict__ Cbf, int ldc) {
    __shared__ __align__(16) unsigned short sA[2][128][40];   // [buf][m][k]
    __shared__ __align__(16) unsigned short sB[128][40];      // [n][k] transposed
    int tid  = threadIdx.x;
    int lane = tid & 31, wid = tid >> 5;
    int m0 = blockIdx.y * 128, n0 = blockIdx.x * 128;
    int mbase = (wid >> 2) * 64;
    int nbase = (wid & 3) * 32;

    const unsigned short* Ab = (const unsigned short*)Aptr;
    const float*          Af = (const float*)Aptr;

    int la   = tid * 8;
    int rowA = la >> 5, colA = la & 31;     // A: 64 rows per round (AMODE 1)
    int kB   = la >> 7, nB   = la & 127;    // B: 16 k-rows per round
    const unsigned short* pB0 = B + (long)kB * ldb + n0 + nB;
    const unsigned short* pB1 = pB0 + (long)16 * ldb;

    const float* pfA0 = Af + (long)(m0 + rowA) * lda + colA;
    const float* pfA1 = pfA0 + (long)64 * lda;
    unsigned short* stA0 = &sA[0][rowA][colA];
    unsigned short* stA1 = &sA[0][rowA + 64][colA];

    unsigned int ldsA = (unsigned int)(unsigned long long)(void*)&sA[0][0][0];

    v8f acc[4][2];
    v8f zero = {};
#pragma unroll
    for (int i = 0; i < 4; ++i)
#pragma unroll
        for (int j = 0; j < 2; ++j) acc[i][j] = zero;

    ushort8 rb0 = *(const ushort8*)pB0;
    ushort8 rb1 = *(const ushort8*)pB1;
    ushort8 ra0, ra1;
    if constexpr (AMODE == 1) {
        ra0 = cvt8(*(const f32x4*)pfA0, *(const f32x4*)(pfA0 + 4));
        ra1 = cvt8(*(const f32x4*)pfA1, *(const f32x4*)(pfA1 + 4));
    } else {
        if (wid == 0) tdm_load_tile(Ab + (long)m0 * lda, lda, ldsA);   // tile 0 -> buf 0
    }

    int khalf = (lane >> 4) * 8;   // A frag: K {0-7,16-23} / {8-15,24-31}
    int kb    = (lane >> 4) * 16;  // unused; B frag reads K via khalf, same order as A
    int buf = 0;

    for (int k0 = 0; k0 < K; k0 += 32) {
        // commit staged B (and A for AMODE 1) to LDS
#pragma unroll
        for (int q = 0; q < 8; ++q) sB[nB + q][kB] = rb0[q];
#pragma unroll
        for (int q = 0; q < 8; ++q) sB[nB + q][kB + 16] = rb1[q];
        if constexpr (AMODE == 1) {
            *(ushort8*)stA0 = ra0;
            *(ushort8*)stA1 = ra1;
        } else {
            if (wid == 0) __builtin_amdgcn_s_wait_tensorcnt(0);
        }
        __syncthreads();
        // issue next tile's loads (fly during the WMMAs below)
        if (k0 + 32 < K) {
            pB0 += (long)32 * ldb;
            pB1 += (long)32 * ldb;
            rb0 = *(const ushort8*)pB0;
            rb1 = *(const ushort8*)pB1;
            __builtin_prefetch(pB0 + (long)32 * ldb, 0, 0);
            if constexpr (AMODE == 1) {
                pfA0 += 32; pfA1 += 32;
                ra0 = cvt8(*(const f32x4*)pfA0, *(const f32x4*)(pfA0 + 4));
                ra1 = cvt8(*(const f32x4*)pfA1, *(const f32x4*)(pfA1 + 4));
            } else {
                if (wid == 0)
                    tdm_load_tile(Ab + (long)m0 * lda + k0 + 32, lda,
                                  ldsA + (unsigned)(buf ^ 1) * ABUFB);
            }
        }
        // fragments per ISA 7.12.2 layouts + 8 WMMAs
        int abuf = (AMODE == 0) ? buf : 0;
        Frag a[4], b[2];
#pragma unroll
        for (int mt = 0; mt < 4; ++mt) {
            int row = mbase + mt * 16 + (lane & 15);
            a[mt].h[0] = *(const ushort8*)&sA[abuf][row][khalf];
            a[mt].h[1] = *(const ushort8*)&sA[abuf][row][16 + khalf];
        }
#pragma unroll
        for (int nt = 0; nt < 2; ++nt) {
            int row = nbase + nt * 16 + (lane & 15);
            b[nt].h[0] = *(const ushort8*)&sB[row][khalf];
            b[nt].h[1] = *(const ushort8*)&sB[row][16 + khalf];
        }
#pragma unroll
        for (int mt = 0; mt < 4; ++mt)
#pragma unroll
            for (int nt = 0; nt < 2; ++nt)
                acc[mt][nt] = __builtin_amdgcn_wmma_f32_16x16x32_bf16(
                    false, a[mt].v, false, b[nt].v, (short)0, acc[mt][nt], false, false);
        buf ^= 1;
        __syncthreads();
    }

    // ---- straight-line epilogue (variant selected at compile time) ----
#pragma unroll
    for (int nt = 0; nt < 2; ++nt) {
        int n = n0 + nbase + nt * 16 + (lane & 15);
        float bv = NGUARD ? ((n < Nbias) ? bias[n] : 0.f) : bias[n];
        long nHW = (long)n * HW_;
#pragma unroll
        for (int mt = 0; mt < 4; ++mt)
#pragma unroll
            for (int vi = 0; vi < 8; ++vi) {
                int m = m0 + mbase + mt * 16 + (lane >> 4) * 8 + vi;
                float v = acc[mt][nt][vi] + bv;
                if (ACT) v = gelu_exact(v);
                if (RES == 1) {
                    v += Res[(long)m * ldres + n];
                } else if (RES == 2) {
                    int bb = m / HW_, rem = m - bb * HW_;
                    v += Res[(long)bb * (CH * HW_) + nHW + rem];
                }
                if (OUT == 0) {
                    Cf[(long)m * ldc + n] = v;
                } else if (OUT == 1) {
                    Cbf[(long)m * ldc + n] = f2bf(v);
                } else {
                    int bb = m / HW_, rem = m - bb * HW_;
                    Cf[(long)bb * (CH * HW_) + nHW + rem] = v;
                }
            }
    }
}

// ---------------------------------------------------------------------------
// Batched mix: Y += Attn(56x56, zero-padded to 64x64) @ U(56x128).
// One workgroup per (b,row). Validity is constant per (wave, lane-half), so
// the accumulate is a single divergence with batched loads then stores.
// wmode 0: h-mix (U rows contiguous); wmode 1: w-mix (U rows strided).
// ---------------------------------------------------------------------------
__global__ __launch_bounds__(256) void k_mix(
    const unsigned short* __restrict__ Attn,   // [1792][64][64] bf16 padded
    const unsigned short* __restrict__ U,      // [tok][128]     bf16
    float* __restrict__ Y,                     // [tok][128]     f32 accumulate
    int wmode) {
    __shared__ __align__(16) unsigned short sU[128][72];   // U^T: [c][j]
    int blk = blockIdx.x;
    int tid = threadIdx.x, lane = tid & 31, wid = tid >> 5;
    long ubase, ustride;
    if (wmode == 0) { ubase = (long)blk * WW * CH; ustride = CH; }
    else { int b = blk / WW, w = blk % WW; ubase = ((long)b * HW_ + w) * CH; ustride = (long)WW * CH; }
    const unsigned short* attn = Attn + (long)blk * 64 * 64;

    // stage U transposed (batch the 4 loads, then scatter)
    ushort8 uv[4];
    int js[4], cs[4];
#pragma unroll
    for (int it = 0; it < 4; ++it) {
        int linear = (it * 256 + tid) * 8;
        int j = linear >> 7, c = linear & 127;
        int jc = (j < 56) ? j : 55;
        js[it] = j; cs[it] = c;
        uv[it] = *(const ushort8*)(U + ubase + (long)jc * ustride + c);
    }
#pragma unroll
    for (int it = 0; it < 4; ++it)
#pragma unroll
        for (int q = 0; q < 8; ++q) sU[cs[it] + q][js[it]] = uv[it][q];
    __syncthreads();

    int mt = wid & 3;        // 4 M tiles of 16
    int ng = wid >> 2;       // 2 N groups of 64
    v8f acc[4];
    v8f zero = {};
#pragma unroll
    for (int i = 0; i < 4; ++i) acc[i] = zero;
    int khalf = (lane >> 4) * 8;
    int kb    = (lane >> 4) * 16;
#pragma unroll
    for (int kk = 0; kk < 2; ++kk) {
        Frag a, b[4];
        int arow = mt * 16 + (lane & 15);
        a.h[0] = *(const ushort8*)(attn + arow * 64 + kk * 32 + khalf);
        a.h[1] = *(const ushort8*)(attn + arow * 64 + kk * 32 + 16 + khalf);
#pragma unroll
        for (int nt = 0; nt < 4; ++nt) {
            int nrow = ng * 64 + nt * 16 + (lane & 15);
            b[nt].h[0] = *(const ushort8*)&sU[nrow][kk * 32 + khalf];
            b[nt].h[1] = *(const ushort8*)&sU[nrow][kk * 32 + 16 + khalf];
        }
#pragma unroll
        for (int nt = 0; nt < 4; ++nt)
            acc[nt] = __builtin_amdgcn_wmma_f32_16x16x32_bf16(
                false, a.v, false, b[nt].v, (short)0, acc[nt], false, false);
    }
    // rows m = mt*16 + half*8 + vi; only (mt==3 && half==1) rows are padding
    if (mt < 3 || (lane >> 4) == 0) {
        float* bp = Y + ubase + (long)(mt * 16 + (lane >> 4) * 8) * ustride
                      + ng * 64 + (lane & 15);
        float r[4][8];
#pragma unroll
        for (int nt = 0; nt < 4; ++nt)
#pragma unroll
            for (int vi = 0; vi < 8; ++vi)
                r[nt][vi] = bp[(long)vi * ustride + nt * 16];
#pragma unroll
        for (int nt = 0; nt < 4; ++nt)
#pragma unroll
            for (int vi = 0; vi < 8; ++vi)
                bp[(long)vi * ustride + nt * 16] = r[nt][vi] + acc[nt][vi];
    }
}

// ---------------------------------------------------------------------------
extern "C" void kernel_launch(void* const* d_in, const int* in_sizes, int n_in,
                              void* d_out, int out_size, void* d_ws, size_t ws_size,
                              hipStream_t stream) {
    (void)in_sizes; (void)n_in; (void)out_size; (void)ws_size;
    const float* x     = (const float*)d_in[0];
    const float* ln1_g = (const float*)d_in[1];
    const float* ln1_b = (const float*)d_in[2];
    const float* h_wnd = (const float*)d_in[3];
    const float* h_bnd = (const float*)d_in[4];
    const float* h_wnn = (const float*)d_in[5];
    const float* h_bnn = (const float*)d_in[6];
    const float* w_wnd = (const float*)d_in[7];
    const float* w_bnd = (const float*)d_in[8];
    const float* w_wnn = (const float*)d_in[9];
    const float* w_bnn = (const float*)d_in[10];
    const float* pc_w  = (const float*)d_in[11];
    const float* pc_b  = (const float*)d_in[12];
    const float* po_w  = (const float*)d_in[13];
    const float* po_b  = (const float*)d_in[14];
    const float* ln2_g = (const float*)d_in[15];
    const float* ln2_b = (const float*)d_in[16];
    const float* fc1_w = (const float*)d_in[17];
    const float* fc1_b = (const float*)d_in[18];
    const float* fc2_w = (const float*)d_in[19];
    const float* fc2_b = (const float*)d_in[20];
    float* out = (float*)d_out;

    char* ws = (char*)d_ws;
    size_t off = 0;
    auto alloc = [&](size_t bytes) -> char* {
        char* p = ws + off;
        off += (bytes + 255) & ~(size_t)255;
        return p;
    };
    unsigned short* u_bf   = (unsigned short*)alloc((size_t)TOK * CH * 2);
    unsigned short* p1h    = (unsigned short*)alloc((size_t)ROWS * KP1P * 2);
    unsigned short* p1w    = (unsigned short*)alloc((size_t)ROWS * KP1P * 2);
    float*          p2h    = (float*)alloc((size_t)ROWS * NP * 4);
    float*          p2w    = (float*)alloc((size_t)ROWS * NP * 4);
    unsigned short* a_h    = (unsigned short*)alloc((size_t)ROWS * 64 * 64 * 2);
    unsigned short* a_w    = (unsigned short*)alloc((size_t)ROWS * 64 * 64 * 2);
    float*          ysum   = (float*)alloc((size_t)TOK * CH * 4);
    float*          xt2    = (float*)alloc((size_t)TOK * CH * 4);
    unsigned short* vbf    = (unsigned short*)alloc((size_t)TOK * CH * 2);
    unsigned short* hbf    = (unsigned short*)alloc((size_t)TOK * EC * 2);
    unsigned short* pcw_bf = (unsigned short*)alloc((size_t)CH * CH * 2);
    unsigned short* pow_bf = (unsigned short*)alloc((size_t)CH * CH * 2);
    unsigned short* fc1_bf = (unsigned short*)alloc((size_t)CH * EC * 2);
    unsigned short* fc2_bf = (unsigned short*)alloc((size_t)EC * CH * 2);
    unsigned short* wnh_bf = (unsigned short*)alloc((size_t)KP1P * NP * 2);
    unsigned short* wnw_bf = (unsigned short*)alloc((size_t)KP1P * NP * 2);

    auto cgrid = [](long n) { return (int)((n + 255) / 256); };

    // zero the padded P1 buffers (tail K columns must be 0)
    k_fill_u16<<<cgrid((long)ROWS * KP1P), 256, 0, stream>>>(p1h, ROWS * KP1P);
    k_fill_u16<<<cgrid((long)ROWS * KP1P), 256, 0, stream>>>(p1w, ROWS * KP1P);

    // weight casts to bf16 (wnn with padding)
    k_cast_bf16<<<cgrid(CH * CH), 256, 0, stream>>>(pc_w, pcw_bf, CH * CH);
    k_cast_bf16<<<cgrid(CH * CH), 256, 0, stream>>>(po_w, pow_bf, CH * CH);
    k_cast_bf16<<<cgrid(CH * EC), 256, 0, stream>>>(fc1_w, fc1_bf, CH * EC);
    k_cast_bf16<<<cgrid(EC * CH), 256, 0, stream>>>(fc2_w, fc2_bf, EC * CH);
    k_cast_pad_wnn<<<cgrid((long)KP1P * NP), 256, 0, stream>>>(h_wnn, wnh_bf);
    k_cast_pad_wnn<<<cgrid((long)KP1P * NP), 256, 0, stream>>>(w_wnn, wnw_bf);

    // LN1 (transposed read from BCHW)
    k_layernorm<<<TOK, 128, 0, stream>>>(x, ln1_g, ln1_b, u_bf, 1);

    // per-token d=10 projections for both mixers
    k_p1<<<cgrid((long)TOK * DD), 256, 0, stream>>>(u_bf, h_wnd, h_bnd, w_wnd, w_bnd, p1h, p1w);

    // dynamic weight generation GEMMs: [1792,576] @ [576,3200] (padded)
    {
        dim3 g(NP / 128, ROWS / 128);
        k_gemm<0, 0, 0, true, 0><<<g, 256, 0, stream>>>(
            p1h, KP1P, wnh_bf, NP, KP1P, NN, h_bnn, nullptr, 0, p2h, nullptr, NP);
        k_gemm<0, 0, 0, true, 0><<<g, 256, 0, stream>>>(
            p1w, KP1P, wnw_bf, NP, KP1P, NN, w_bnn, nullptr, 0, p2w, nullptr, NP);
    }
    // softmax -> padded bf16 attention [1792][64][64]
    k_softmax56<<<ROWS * 64, 64, 0, stream>>>(p2h, a_h);
    k_softmax56<<<ROWS * 64, 64, 0, stream>>>(p2w, a_w);

    // y_c = u @ pc_w + pc_b   (overwrites ysum)
    {
        dim3 g(1, TOK / 128);
        k_gemm<0, 0, 0, false, 0><<<g, 256, 0, stream>>>(
            u_bf, CH, pcw_bf, CH, CH, CH, pc_b, nullptr, 0, ysum, nullptr, CH);
    }
    // ysum += y_h ; ysum += y_w
    k_mix<<<ROWS, 256, 0, stream>>>(a_h, u_bf, ysum, 0);
    k_mix<<<ROWS, 256, 0, stream>>>(a_w, u_bf, ysum, 1);

    // proj_o (A = ysum f32, converted in-kernel) + residual(x transposed) -> xt2
    {
        dim3 g(1, TOK / 128);
        k_gemm<2, 0, 0, false, 1><<<g, 256, 0, stream>>>(
            ysum, CH, pow_bf, CH, CH, CH, po_b, x, 0, xt2, nullptr, CH);
    }
    // LN2 -> vbf
    k_layernorm<<<TOK, 128, 0, stream>>>(xt2, ln2_g, ln2_b, vbf, 0);
    // fc1 + GELU -> hbf (bf16)
    {
        dim3 g(EC / 128, TOK / 128);
        k_gemm<0, 1, 1, false, 0><<<g, 256, 0, stream>>>(
            vbf, CH, fc1_bf, EC, CH, EC, fc1_b, nullptr, 0, nullptr, hbf, EC);
    }
    // fc2 + residual(xt2) -> out (transposed store to BCHW)
    {
        dim3 g(1, TOK / 128);
        k_gemm<1, 2, 0, false, 0><<<g, 256, 0, stream>>>(
            hbf, EC, fc2_bf, CH, EC, CH, fc2_b, xt2, CH, out, nullptr, CH);
    }
}
